// DSCLRCN_86878598463629
// MI455X (gfx1250) — matrix operand order repaired and checked by reading, not compiled, not run on hardware
//
#include <hip/hip_runtime.h>
#include <hip/hip_bf16.h>
#include <cstdint>

#define HIDN 128
#define GHh  60
#define GWw  80
#define NBat 8
#define INH  480
#define INW  640

typedef __attribute__((ext_vector_type(16))) __bf16 v16bf;
typedef __attribute__((ext_vector_type(8)))  __bf16 v8bf;
typedef __attribute__((ext_vector_type(4)))  __bf16 v4bf;
typedef __attribute__((ext_vector_type(8)))  float  v8f;

// ---------------------------------------------------------------- helpers ---
static __device__ __forceinline__ float sigmoidf_(float x) {
  return 1.0f / (1.0f + __expf(-x));
}

// A fragment: 16x32 bf16, lane (0..15) = row M, contiguous K layout:
// hi=0 lanes hold K 0..15; hi=1 lanes hold K 16..31
static __device__ __forceinline__ v16bf load_a_frag(const __bf16* row_base, int k0, int hi) {
  const __bf16* q = row_base + k0 + hi * 16;
  v8bf a = *(const v8bf*)q;
  v8bf b = *(const v8bf*)(q + 8);
  v16bf f;
#pragma unroll
  for (int i = 0; i < 8; ++i) { f[i] = a[i]; f[i + 8] = b[i]; }
  return f;
}

// B fragment: 32x16 bf16, lane (0..15) = col N, per-lane contiguous K range
// (lanes 0-15: K 0..15, lanes 16-31: K 16..31) -> single 32B load
static __device__ __forceinline__ v16bf load_b_frag(const __bf16* col_base, int k0, int hi) {
  return *(const v16bf*)(col_base + k0 + hi * 16);
}

// ---------------------------------------------------------------- prep ------
// n must be a multiple of 4 (all our tensors are)
__global__ void k_f32_to_bf16(const float* __restrict__ in, __bf16* __restrict__ out, int n4) {
  int i = blockIdx.x * 256 + threadIdx.x;
  if (i >= n4) return;
  float4 v = ((const float4*)in)[i];
  v4bf b;
  b[0] = (__bf16)v.x; b[1] = (__bf16)v.y; b[2] = (__bf16)v.z; b[3] = (__bf16)v.w;
  ((v4bf*)out)[i] = b;
}

__global__ void k_bias_sum(const float* __restrict__ a, const float* __restrict__ b,
                           float* __restrict__ o, int n) {
  int i = blockIdx.x * 256 + threadIdx.x;
  if (i < n) o[i] = a[i] + b[i];
}

// lf [8,512,60,80] -> featIn [8,60,80,512]
__global__ void k_nchw_to_nhwc(const float* __restrict__ in, float* __restrict__ out, int n) {
  int idx = blockIdx.x * 256 + threadIdx.x;
  if (idx >= n) return;
  int d = idx & 511;
  int p = idx >> 9;
  int x = p % GWw;
  int y = (p / GWw) % GHh;
  int nn = p / (GWw * GHh);
  out[idx] = in[(((size_t)nn * 512 + d) * GHh + y) * GWw + x];
}

// out[n,o] = dot(ctx[n,:128], W[o,:128]) + b[o]
__global__ void k_fc(const float* __restrict__ ctx, const float* __restrict__ W,
                     const float* __restrict__ b, float* __restrict__ out, int O) {
  int i = blockIdx.x * 256 + threadIdx.x;
  if (i >= NBat * O) return;
  int n = i / O, o = i % O;
  const float4* c4 = (const float4*)(ctx + (size_t)n * HIDN);
  const float4* w4 = (const float4*)(W + (size_t)o * HIDN);
  float s = 0.f;
#pragma unroll
  for (int k = 0; k < HIDN / 4; ++k) {
    float4 a = c4[k], w = w4[k];
    s += a.x * w.x + a.y * w.y + a.z * w.z + a.w * w.w;
  }
  out[i] = s + b[o];
}

// ------------------------------------------------------- gx = x*wih^T + bias
// gx layout: [2 dirs][S*T][512].  Rows r map to (s=r/T, t=r%T); t==0 / t==T-1
// use the context vector for image n = s/div.  blockIdx.y = direction.
// Whole 16 x Din A-tile staged to LDS as bf16 once; K-loop is barrier-free.
// Din is a template constant so all tile strides fold into load offsets.
template <int Din>
__global__ __launch_bounds__(256) void k_gx_gemm(
    const float* __restrict__ X, const float* __restrict__ C,
    const __bf16* __restrict__ wih, const float* __restrict__ bias,
    float* __restrict__ gx,
    int S, int T, int div,
    long long xA, long long xB, long long xT)
{
  extern __shared__ char smem[];
  __bf16* a_lds = (__bf16*)smem;                 // [16][Din+8] bf16 (padded)
  constexpr int lda = Din + 8;
  const int tid  = threadIdx.x;
  const int wave = tid >> 5, lane = tid & 31, lo = lane & 15, hi = lane >> 4;
  const int r0   = blockIdx.x * 16;
  const int dir  = blockIdx.y;

  // ---- stage full A tile (16 rows x Din) to LDS as bf16, one barrier ----
  constexpr int nv4 = Din >> 2;
#pragma unroll 2
  for (int idx = tid; idx < 16 * nv4; idx += 256) {
    const int row = idx / nv4;
    const int c4  = (idx % nv4) * 4;
    const int rg = r0 + row;
    const int s = rg / T, t = rg % T;
    const float* xr;
    if (t == 0 || t == T - 1) xr = C + (size_t)(s / div) * Din;
    else xr = X + (size_t)(s / div) * xA + (size_t)(s % div) * xB + (size_t)(t - 1) * xT;
    float4 v = *(const float4*)(xr + c4);
    v4bf b;
    b[0] = (__bf16)v.x; b[1] = (__bf16)v.y; b[2] = (__bf16)v.z; b[3] = (__bf16)v.w;
    *(v4bf*)(a_lds + row * lda + c4) = b;
  }
  __syncthreads();

  // ---- K loop: all loads issued before the 4 WMMAs (latency overlap) ----
  const __bf16* bbase = wih + ((size_t)dir * 512 + wave * 64 + lo) * Din;

  v8f acc[4] = {{}, {}, {}, {}};
  constexpr int nk = Din >> 5;
#pragma unroll 2
  for (int kc = 0; kc < nk; ++kc) {
    const int k0 = kc * 32;
    v16bf af = load_a_frag(a_lds + lo * lda, k0, hi);
    v16bf bf[4];
#pragma unroll
    for (int i = 0; i < 4; ++i)
      bf[i] = load_b_frag(bbase + i * 16 * Din, k0, hi);   // const offsets
#pragma unroll
    for (int i = 0; i < 4; ++i)
      acc[i] = __builtin_amdgcn_wmma_f32_16x16x32_bf16(false, af, false, bf[i], (short)0, acc[i], false, false);
  }

  // ---- store with bias; C/D layout: lane -> col, element r -> row (r+hi*8) ----
#pragma unroll
  for (int i = 0; i < 4; ++i) {
    const int gg = wave * 64 + i * 16 + lo;
    const float bv = bias[dir * 512 + gg];
#pragma unroll
    for (int r = 0; r < 8; ++r) {
      const int m = r + hi * 8;
      gx[((size_t)dir * S * T + (r0 + m)) * 512 + gg] = acc[i][r] + bv;
    }
  }
}

// ----------------------------------------------------------- LSTM scan ------
// One workgroup = 16 sequences x 1 direction, persistent over T steps.
// whh B-fragments preloaded in VGPRs; h in LDS (bf16); c in thread registers.
// Next timestep's gx block is prefetched each step (global_prefetch_b8).
__global__ __launch_bounds__(256, 1) void k_lstm_scan(
    const float* __restrict__ gx,      // [2][S*T][512]
    const __bf16* __restrict__ whh,    // [2][512][128]
    float* __restrict__ out,
    int S, int T, int div,
    long long oA, long long oB, long long oT)
{
  extern __shared__ char smem[];
  __bf16* h_lds = (__bf16*)smem;                      // [16][128] bf16
  float*  gates = (float*)(smem + 16 * 128 * 2);      // [16][516] f32 (padded)

  const int tid  = threadIdx.x;
  const int wave = tid >> 5, lane = tid & 31, lo = lane & 15, hi = lane >> 4;
  const int dir = blockIdx.y;
  const int s0  = blockIdx.x * 16;
  const size_t gx0 = (size_t)dir * S * T * 512;
  const __bf16* whh_d = whh + (size_t)dir * 512 * 128;

  // Preload whh^T fragments: wave owns gate tiles [wave*64, wave*64+64)
  v16bf bfrag[4][4];
#pragma unroll
  for (int i = 0; i < 4; ++i) {
    const int g0 = wave * 64 + i * 16;
#pragma unroll
    for (int kc = 0; kc < 4; ++kc)
      bfrag[i][kc] = load_b_frag(whh_d + (size_t)(g0 + lo) * 128, kc * 32, hi);
  }

  // init h = 0 (bf16), c = 0 (regs)
  for (int i = tid; i < 16 * 128; i += 256) h_lds[i] = __builtin_bit_cast(__bf16, (unsigned short)0);
  float creg[8];
#pragma unroll
  for (int k = 0; k < 8; ++k) creg[k] = 0.f;
  const int jh   = tid & 127;   // hidden unit this thread owns
  const int seqb = tid >> 7;    // 0/1; thread covers seqs seqb, seqb+2, ..., seqb+14
  __syncthreads();

  for (int step = 0; step < T; ++step) {
    const int t = dir ? (T - 1 - step) : step;

    // prefetch next step's gx block: 256 threads cover 16 rows x 512 cols (f32)
    if (step + 1 < T) {
      const int tn = dir ? (T - 2 - step) : (step + 1);
      const float* pf = gx + gx0 + ((size_t)(s0 + (tid & 15)) * T + tn) * 512 + (tid >> 4) * 32;
      __builtin_prefetch(pf, 0, 1);
    }

    // issue ALL loads (A frags from LDS + 32 gx accumulator-init loads) first
    v16bf af[4];
#pragma unroll
    for (int kc = 0; kc < 4; ++kc) af[kc] = load_a_frag(h_lds + lo * 128, kc * 32, hi);

    v8f acc[4];
#pragma unroll
    for (int i = 0; i < 4; ++i) {
      const int gcol = wave * 64 + i * 16 + lo;
      const float* grow = gx + gx0 + ((size_t)s0 * T + t) * 512 + gcol;
#pragma unroll
      for (int r = 0; r < 8; ++r) {
        const int m = r + hi * 8;
        acc[i][r] = grow[(size_t)m * T * 512];
      }
    }

    // 4 independent WMMA accumulation chains
#pragma unroll
    for (int i = 0; i < 4; ++i) {
#pragma unroll
      for (int kc = 0; kc < 4; ++kc)
        acc[i] = __builtin_amdgcn_wmma_f32_16x16x32_bf16(false, af[kc], false, bfrag[i][kc], (short)0, acc[i], false, false);
    }

#pragma unroll
    for (int i = 0; i < 4; ++i) {
      const int gcol = wave * 64 + i * 16 + lo;
#pragma unroll
      for (int r = 0; r < 8; ++r) {
        const int m = r + hi * 8;
        gates[m * 516 + gcol] = acc[i][r];
      }
    }
    __syncthreads();

    const bool interior = (t >= 1) && (t <= T - 2);   // uniform across block
#pragma unroll
    for (int k = 0; k < 8; ++k) {
      const int seq = seqb + 2 * k;
      float gi = gates[seq * 516 + jh];
      float gf = gates[seq * 516 + 128 + jh];
      float gg = gates[seq * 516 + 256 + jh];
      float go = gates[seq * 516 + 384 + jh];
      float c = sigmoidf_(gf) * creg[k] + sigmoidf_(gi) * tanhf(gg);
      float h = sigmoidf_(go) * tanhf(c);
      creg[k] = c;
      h_lds[seq * 128 + jh] = (__bf16)h;
      if (interior) {
        int sg = s0 + seq;
        size_t offo = (size_t)(sg / div) * oA + (size_t)(sg % div) * oB +
                      (size_t)(t - 1) * oT + dir * 128 + jh;
        out[offo] = h;
      }
    }
    __syncthreads();
  }
}

// ---------------------------------------------------------------- tail ------
__global__ void k_conv1(const float* __restrict__ f, const float* __restrict__ w,
                        const float* __restrict__ b, float* __restrict__ o, int P) {
  int p = blockIdx.x * 256 + threadIdx.x;
  if (p >= P) return;
  const float4* fr = (const float4*)(f + (size_t)p * 256);
  const float4* wr = (const float4*)w;
  float s = 0.f;
#pragma unroll
  for (int i = 0; i < 64; ++i) {
    float4 a = fr[i], ww = wr[i];
    s += a.x * ww.x + a.y * ww.y + a.z * ww.z + a.w * ww.w;
  }
  o[p] = s + b[0];
}

__global__ void k_imgmax(const float* __restrict__ logits, float* __restrict__ m) {
  const int n = blockIdx.x;
  float mx = -3.402823466e38f;
  for (int i = threadIdx.x; i < GHh * GWw; i += 256) mx = fmaxf(mx, logits[n * GHh * GWw + i]);
  __shared__ float red[256];
  red[threadIdx.x] = mx; __syncthreads();
  for (int s = 128; s > 0; s >>= 1) {
    if (threadIdx.x < s) red[threadIdx.x] = fmaxf(red[threadIdx.x], red[threadIdx.x + s]);
    __syncthreads();
  }
  if (threadIdx.x == 0) m[n] = red[0];
}

__global__ void k_zero8(float* __restrict__ sums) {
  if (threadIdx.x < NBat) sums[threadIdx.x] = 0.f;
}

// align_corners bilinear upsample + exp(x - max) + per-image partial sums
__global__ void k_resize_exp(const float* __restrict__ logits, const float* __restrict__ mbuf,
                             float* __restrict__ outv, float* __restrict__ sums) {
  const int n = blockIdx.y;
  const int p = blockIdx.x * 256 + threadIdx.x;
  const int Y = p / INW, X = p % INW;
  float py = Y * ((float)(GHh - 1) / (float)(INH - 1));
  float px = X * ((float)(GWw - 1) / (float)(INW - 1));
  int y0 = min((int)floorf(py), GHh - 2);
  int x0 = min((int)floorf(px), GWw - 2);
  float fy = py - (float)y0, fx = px - (float)x0;
  const float* L = logits + (size_t)n * GHh * GWw;
  float v00 = L[y0 * GWw + x0],       v01 = L[y0 * GWw + x0 + 1];
  float v10 = L[(y0 + 1) * GWw + x0], v11 = L[(y0 + 1) * GWw + x0 + 1];
  float v = (v00 * (1.f - fy) + v10 * fy) * (1.f - fx) + (v01 * (1.f - fy) + v11 * fy) * fx;
  float e = __expf(v - mbuf[n]);
  outv[(size_t)n * INH * INW + p] = e;
  __shared__ float red[256];
  red[threadIdx.x] = e; __syncthreads();
  for (int s = 128; s > 0; s >>= 1) {
    if (threadIdx.x < s) red[threadIdx.x] += red[threadIdx.x + s];
    __syncthreads();
  }
  if (threadIdx.x == 0) atomicAdd(&sums[n], red[0]);
}

__global__ void k_norm(float* __restrict__ outv, const float* __restrict__ sums) {
  const int n = blockIdx.y;
  const int p = blockIdx.x * 256 + threadIdx.x;
  outv[(size_t)n * INH * INW + p] *= (1.0f / sums[n]);
}

// ---------------------------------------------------------------- host ------
extern "C" void kernel_launch(void* const* d_in, const int* in_sizes, int n_in,
                              void* d_out, int out_size, void* d_ws, size_t ws_size,
                              hipStream_t stream) {
  (void)in_sizes; (void)n_in; (void)out_size; (void)ws_size;
  const float* lf     = (const float*)d_in[0];
  const float* ctx    = (const float*)d_in[1];
  const float* fc1_w  = (const float*)d_in[2];
  const float* fc1_b  = (const float*)d_in[3];
  const float* fcr_w  = (const float*)d_in[4];
  const float* fcr_b  = (const float*)d_in[5];
  const float* conv_w = (const float*)d_in[6];
  const float* conv_b = (const float*)d_in[7];
  float* out = (float*)d_out;

  char* base = (char*)d_ws;
  size_t off = 0;
  auto take = [&](size_t bytes) -> char* {
    char* p = base + off;
    off += (bytes + 255) & ~(size_t)255;
    return p;
  };

  float* gx     = (float*)take((size_t)2 * 640 * 62 * 512 * 4);   // max over passes
  float* featIn = (float*)take((size_t)NBat * GHh * GWw * 512 * 4);
  float* bufA   = (float*)take((size_t)NBat * GHh * GWw * 256 * 4);
  float* bufB   = (float*)take((size_t)NBat * GHh * GWw * 256 * 4);
  float* c1     = (float*)take((size_t)NBat * 512 * 4);
  float* cr     = (float*)take((size_t)NBat * 256 * 4);
  float* logits = (float*)take((size_t)NBat * GHh * GWw * 4);
  float* mbuf   = (float*)take(NBat * 4);
  float* sums   = (float*)take(NBat * 4);

  const int din[4] = {512, 256, 256, 256};
  __bf16* wih_bf[4]; __bf16* whh_bf[4]; float* bias[4];
  for (int p = 0; p < 4; ++p) {
    wih_bf[p] = (__bf16*)take((size_t)2 * 512 * din[p] * 2);
    whh_bf[p] = (__bf16*)take((size_t)2 * 512 * 128 * 2);
    bias[p]   = (float*)take((size_t)2 * 512 * 4);
  }

  // --- weight prep (bf16 conversion + bias fusion) ---
  for (int p = 0; p < 4; ++p) {
    const int wi = 8 + p * 4;
    const int nwih4 = 2 * 512 * din[p] / 4;
    k_f32_to_bf16<<<(nwih4 + 255) / 256, 256, 0, stream>>>((const float*)d_in[wi], wih_bf[p], nwih4);
    const int nwhh4 = 2 * 512 * 128 / 4;
    k_f32_to_bf16<<<(nwhh4 + 255) / 256, 256, 0, stream>>>((const float*)d_in[wi + 1], whh_bf[p], nwhh4);
    k_bias_sum<<<4, 256, 0, stream>>>((const float*)d_in[wi + 2], (const float*)d_in[wi + 3], bias[p], 1024);
  }

  // --- lf NCHW -> NHWC ---
  {
    const int n = NBat * GHh * GWw * 512;
    k_nchw_to_nhwc<<<(n + 255) / 256, 256, 0, stream>>>(lf, featIn, n);
  }

  // --- context projections ---
  k_fc<<<(NBat * 512 + 255) / 256, 256, 0, stream>>>(ctx, fc1_w, fc1_b, c1, 512);
  k_fc<<<(NBat * 256 + 255) / 256, 256, 0, stream>>>(ctx, fcr_w, fcr_b, cr, 256);

  // --- four BiLSTM passes (row, col, row, col), ping-pong buffers ---
  const float* passIn[4]  = {featIn, bufA, bufB, bufA};
  float*       passOut[4] = {bufA, bufB, bufA, bufB};
  const float* passCtx[4] = {c1, cr, cr, cr};
  const int    isRow[4]   = {1, 0, 1, 0};

  for (int p = 0; p < 4; ++p) {
    int S, L, div; long long xA, xB, xT, oA, oB, oT;
    if (isRow[p]) {
      S = NBat * GHh; L = GWw; div = GHh;
      xT = din[p]; xB = (long long)GWw * din[p]; xA = (long long)GHh * GWw * din[p];
      oA = (long long)GHh * GWw * 256; oB = (long long)GWw * 256; oT = 256;
    } else {
      S = NBat * GWw; L = GHh; div = GWw;
      xT = (long long)GWw * 256; xB = 256; xA = (long long)GHh * GWw * 256;
      oA = (long long)GHh * GWw * 256; oB = 256; oT = (long long)GWw * 256;
    }
    const int T = L + 2;
    dim3 ggrid((S * T) / 16, 2);                       // blockIdx.y = direction
    const size_t gemm_lds = (size_t)16 * (din[p] + 8) * 2;
    if (din[p] == 512)
      k_gx_gemm<512><<<ggrid, 256, gemm_lds, stream>>>(passIn[p], passCtx[p], wih_bf[p], bias[p],
                                                       gx, S, T, div, xA, xB, xT);
    else
      k_gx_gemm<256><<<ggrid, 256, gemm_lds, stream>>>(passIn[p], passCtx[p], wih_bf[p], bias[p],
                                                       gx, S, T, div, xA, xB, xT);
    dim3 sgrid(S / 16, 2);
    const size_t scan_lds = 16 * 128 * 2 + 16 * 516 * 4;   // h(bf16) + gates(f32)
    k_lstm_scan<<<sgrid, 256, scan_lds, stream>>>(gx, whh_bf[p], passOut[p], S, T, div, oA, oB, oT);
  }

  // --- 1x1 conv -> bilinear upsample -> softmax ---
  k_conv1<<<(NBat * GHh * GWw + 255) / 256, 256, 0, stream>>>(bufB, conv_w, conv_b, logits, NBat * GHh * GWw);
  k_imgmax<<<NBat, 256, 0, stream>>>(logits, mbuf);
  k_zero8<<<1, 32, 0, stream>>>(sums);
  dim3 rgrid(INH * INW / 256, NBat);
  k_resize_exp<<<rgrid, 256, 0, stream>>>(logits, mbuf, out, sums);
  k_norm<<<rgrid, 256, 0, stream>>>(out, sums);
}
